// GPT2_60782377173490
// MI455X (gfx1250) — matrix operand order repaired and checked by reading, not compiled, not run on hardware
//
#include <hip/hip_runtime.h>
#include <hip/hip_bf16.h>
#include <math.h>

// ---------------------------------------------------------------------------
// GPT-2 small forward for MI455X (gfx1250, wave32, WMMA).
// All GEMMs use v_wmma_f32_16x16x32_bf16 (bf16 A/B, f32 accum).
// ---------------------------------------------------------------------------

typedef __bf16 bf16;
typedef __attribute__((ext_vector_type(16))) __bf16 v16bf;
typedef __attribute__((ext_vector_type(8)))  __bf16 v8bf;
typedef __attribute__((ext_vector_type(8)))  float  v8f;

constexpr int NB  = 4;         // batch
constexpr int NT  = 1024;      // seq len
constexpr int NC  = 768;       // channels
constexpr int NH  = 12;        // heads
constexpr int NHS = 64;        // head size
constexpr int NL  = 12;        // layers
constexpr int NV  = 50257;     // vocab
constexpr int NC3 = 3 * NC;
constexpr int NC4 = 4 * NC;
constexpr int NBT = NB * NT;   // 4096 rows
constexpr float ATT_SCALE = 0.125f;   // 1/sqrt(64)
constexpr float NEG_BIG   = -1.0e30f; // finite "-inf" (exp underflows to 0)

__device__ __forceinline__ bf16 f2bf(float f) {
    union { float f; unsigned u; } v; v.f = f;
    unsigned r = v.u + 0x7FFFu + ((v.u >> 16) & 1u);   // round-to-nearest-even
    unsigned short h = (unsigned short)(r >> 16);
    bf16 o; __builtin_memcpy(&o, &h, 2);
    return o;
}

__device__ __forceinline__ v16bf mk16(v8bf lo, v8bf hi) {
    return __builtin_shufflevector(lo, hi, 0,1,2,3,4,5,6,7,8,9,10,11,12,13,14,15);
}

// ---------------------------------------------------------------------------
// fp32 -> bf16 bulk convert (weights; grid-stride)
// ---------------------------------------------------------------------------
__global__ void __launch_bounds__(256)
k_cvt_bf16(const float* __restrict__ src, bf16* __restrict__ dst, size_t n) {
    size_t i = (size_t)blockIdx.x * blockDim.x + threadIdx.x;
    size_t stride = (size_t)gridDim.x * blockDim.x;
    for (; i < n; i += stride) dst[i] = f2bf(src[i]);
}

// ---------------------------------------------------------------------------
// x[b,t,:] = wte[inp[b,t],:] + wpe[t,:]
// ---------------------------------------------------------------------------
__global__ void __launch_bounds__(256)
k_embed(const int* __restrict__ inp, const float* __restrict__ wte,
        const float* __restrict__ wpe, float* __restrict__ x) {
    const int row = blockIdx.x;            // b*T + t
    const int t   = row % NT;
    const int tok = inp[row];
    const float* we = wte + (size_t)tok * NC;
    const float* pe = wpe + (size_t)t   * NC;
    float* xr = x + (size_t)row * NC;
    for (int i = threadIdx.x; i < NC; i += 256) xr[i] = we[i] + pe[i];
}

// ---------------------------------------------------------------------------
// LayerNorm: one block per row; f32 stats, bf16 output (feeds WMMA A-operand)
// ---------------------------------------------------------------------------
__global__ void __launch_bounds__(256)
k_layernorm(const float* __restrict__ x, const float* __restrict__ g,
            const float* __restrict__ bta, bf16* __restrict__ out) {
    __shared__ float red[256];
    const int row = blockIdx.x, tid = threadIdx.x;
    const float* xr = x + (size_t)row * NC;

    float s = 0.0f;
    for (int i = tid; i < NC; i += 256) s += xr[i];
    red[tid] = s; __syncthreads();
    for (int st = 128; st > 0; st >>= 1) {
        if (tid < st) red[tid] += red[tid + st];
        __syncthreads();
    }
    const float mu = red[0] * (1.0f / NC);
    __syncthreads();

    float v = 0.0f;
    for (int i = tid; i < NC; i += 256) { float d = xr[i] - mu; v += d * d; }
    red[tid] = v; __syncthreads();
    for (int st = 128; st > 0; st >>= 1) {
        if (tid < st) red[tid] += red[tid + st];
        __syncthreads();
    }
    const float rstd = rsqrtf(red[0] * (1.0f / NC) + 1e-5f);

    bf16* orow = out + (size_t)row * NC;
    for (int i = tid; i < NC; i += 256)
        orow[i] = f2bf((xr[i] - mu) * rstd * g[i] + bta[i]);
}

// ---------------------------------------------------------------------------
// Tiled bf16 GEMM: out(MxN) = A(MxK) * B + bias [+resid] [GELU] ; f32 accum.
//   TRANSB=0: B is row-major KxN.  TRANSB=1: B is row-major NxK (weight-tied head).
//   M assumed multiple of 128, K multiple of 32; N bounds-checked.
// Block: 256 threads = 8 waves (2x4), block tile 128x128, K-step 32.
// Wave tile 64x32 -> 4x2 wmma_f32_16x16x32_bf16 accumulators.
// ---------------------------------------------------------------------------
template<bool TRANSB, bool DO_GELU, bool RESID, bool OUTBF>
__global__ void __launch_bounds__(256)
k_gemm(const bf16* __restrict__ A, const bf16* __restrict__ Bm,
       const float* __restrict__ bias, const float* __restrict__ resid,
       void* __restrict__ outp, int M, int N, int K) {
    constexpr int BM = 128, BN = 128, BK = 32, LDK = BK + 8;   // +8 bf16 pad
    __shared__ bf16 As[BM][LDK];   // [m][k]  (A-fragment layout)
    __shared__ bf16 Bs[BN][LDK];   // [n][k]  (B-fragment layout)

    const int tid  = threadIdx.x;
    const int lane = tid & 31;
    const int wave = tid >> 5;
    const int wm   = (wave >> 2) * 64;     // 0 / 64
    const int wn   = (wave & 3)  * 32;     // 0..96
    const int l15  = lane & 15;
    const int kh   = lane >> 4;            // half-wave select
    const int m0   = blockIdx.y * BM;
    const int n0   = blockIdx.x * BN;

    v8f acc[4][2] = {};

    // staging thread maps
    const int arow = tid >> 1, ac0 = (tid & 1) * 16;           // A: 16 bf16 contiguous
    const int bkk  = tid >> 3, bn0 = (tid & 7) * 16;           // B (KxN): row k, 16 along n
    const int tbn  = tid >> 1, tbc0 = (tid & 1) * 16;          // B^T (NxK): row n, 16 along k

    for (int k0 = 0; k0 < K; k0 += BK) {
        // ---- stage A tile (always in-bounds) ----
        {
            const bf16* src = A + (size_t)(m0 + arow) * K + k0 + ac0;
            *(v8bf*)&As[arow][ac0]     = *(const v8bf*)(src);
            *(v8bf*)&As[arow][ac0 + 8] = *(const v8bf*)(src + 8);
        }
        // ---- stage B tile into [n][k] layout ----
        if (!TRANSB) {
            const bf16* src = Bm + (size_t)(k0 + bkk) * N + n0 + bn0;
            #pragma unroll
            for (int j = 0; j < 16; ++j)
                Bs[bn0 + j][bkk] = (n0 + bn0 + j < N) ? src[j] : f2bf(0.0f);
        } else {
            const int n = n0 + tbn;
            if (n < N) {
                const bf16* src = Bm + (size_t)n * K + k0 + tbc0;
                *(v8bf*)&Bs[tbn][tbc0]     = *(const v8bf*)(src);
                *(v8bf*)&Bs[tbn][tbc0 + 8] = *(const v8bf*)(src + 8);
            } else {
                #pragma unroll
                for (int j = 0; j < 16; ++j) Bs[tbn][tbc0 + j] = f2bf(0.0f);
            }
        }
        __syncthreads();

        if (k0 + BK < K) {  // hint next A tile into cache (global_prefetch_b8)
            __builtin_prefetch(A + (size_t)(m0 + arow) * K + k0 + BK + ac0, 0, 0);
        }

        // ---- fragments + WMMA ----
        v16bf afr[4];
        #pragma unroll
        for (int mt = 0; mt < 4; ++mt) {
            const bf16* ap = &As[wm + mt * 16 + l15][0];
            v8bf lo = *(const v8bf*)(ap + kh * 8);          // K = kh*8 .. +8
            v8bf hi = *(const v8bf*)(ap + 16 + kh * 8);     // K = 16+kh*8 .. +8
            afr[mt] = mk16(lo, hi);
        }
        #pragma unroll
        for (int nt = 0; nt < 2; ++nt) {
            const bf16* bp = &Bs[wn + nt * 16 + l15][0];  // K = kh*8 .. +8, 16+kh*8 .. +8
            v8bf lo = *(const v8bf*)(bp + kh * 8);
            v8bf hi = *(const v8bf*)(bp + 16 + kh * 8);
            v16bf bfr = mk16(lo, hi);
            #pragma unroll
            for (int mt = 0; mt < 4; ++mt)
                acc[mt][nt] = __builtin_amdgcn_wmma_f32_16x16x32_bf16(
                    false, afr[mt], false, bfr, (short)0, acc[mt][nt], false, false);
        }
        __syncthreads();
    }

    // ---- epilogue: C layout VGPR r -> row r + 8*kh, col lane&15 ----
    #pragma unroll
    for (int mt = 0; mt < 4; ++mt)
    #pragma unroll
    for (int nt = 0; nt < 2; ++nt)
    #pragma unroll
    for (int r = 0; r < 8; ++r) {
        const int row = m0 + wm + mt * 16 + r + 8 * kh;
        const int col = n0 + wn + nt * 16 + l15;
        if (col < N) {
            float v = acc[mt][nt][r];
            if (bias)  v += bias[col];
            if (RESID) v += resid[(size_t)row * N + col];
            if (DO_GELU) v = 0.5f * v * (1.0f + erff(v * 0.70710678118654752f));
            if (OUTBF) ((bf16*)outp)[(size_t)row * N + col] = f2bf(v);
            else       ((float*)outp)[(size_t)row * N + col] = v;
        }
    }
}

// ---------------------------------------------------------------------------
// Fused causal attention (flash style), one block per (b, h, 64-row Q tile).
// 128 threads = 4 waves; each wave owns 16 query rows.
// Q pre-scaled at stage time; online softmax; both GEMMs on WMMA bf16.
// qkv layout: (B,T,H,3*HS) flattened -> per (b,t,h): q @ +0, k @ +HS, v @ +2HS.
// ---------------------------------------------------------------------------
__global__ void __launch_bounds__(128)
k_attention(const float* __restrict__ qkv, bf16* __restrict__ atty) {
    constexpr int BQ = 64, BKV = 64, LD = NHS + 8;   // 72 bf16 rows (144B, 16B aligned)
    __shared__ bf16 Qs[BQ][LD];    // [q][d]  A-layout for Q*K^T
    __shared__ bf16 Ks[BKV][LD];   // [key][d] == B-layout [n][k] for Q*K^T
    __shared__ bf16 Vs[NHS][LD];   // [d][key] == B-layout [n][k] for P*V
    __shared__ bf16 Ps[BQ][LD];    // [q][key] A-layout for P*V

    const int qb = blockIdx.x, h = blockIdx.y, b = blockIdx.z;
    const int tid = threadIdx.x, lane = tid & 31, wave = tid >> 5;
    const int l15 = lane & 15, kh = lane >> 4;
    const int wq  = wave * 16;                       // wave's local q-row base
    const int qstride = 3 * NC;                      // fp32 elems per (b,t)
    const int hbase   = h * 3 * NHS;

    // ---- stage Q (scaled) ----
    {
        const int row = tid >> 1, c0 = (tid & 1) * 32;
        const int tq = qb * BQ + row;
        const float* src = qkv + (size_t)(b * NT + tq) * qstride + hbase;
        #pragma unroll
        for (int j = 0; j < 32; ++j)
            Qs[row][c0 + j] = f2bf(src[c0 + j] * ATT_SCALE);
    }

    float mstat[8], lstat[8];
    #pragma unroll
    for (int r = 0; r < 8; ++r) { mstat[r] = NEG_BIG; lstat[r] = 0.0f; }
    v8f o[4] = {};

    for (int jb = 0; jb <= qb; ++jb) {               // causal block pruning
        const int kb0 = jb * BKV;
        // ---- stage K tile [key][d], V tile transposed [d][key] ----
        {
            const int row = tid >> 1, c0 = (tid & 1) * 32;
            const int tk = kb0 + row;
            const float* kp = qkv + (size_t)(b * NT + tk) * qstride + hbase + NHS;
            const float* vp = kp + NHS;
            #pragma unroll
            for (int j = 0; j < 32; ++j) Ks[row][c0 + j] = f2bf(kp[c0 + j]);
            #pragma unroll
            for (int j = 0; j < 32; ++j) Vs[c0 + j][row] = f2bf(vp[c0 + j]);
        }
        __syncthreads();

        // ---- S = (Q*scale) @ K^T : 4 key tiles x 2 K-steps ----
        v8f s[4] = {};
        #pragma unroll
        for (int nt = 0; nt < 4; ++nt) {
            #pragma unroll
            for (int kk = 0; kk < 2; ++kk) {
                const bf16* ap = &Qs[wq + l15][kk * 32];
                v16bf a = mk16(*(const v8bf*)(ap + kh * 8),
                               *(const v8bf*)(ap + 16 + kh * 8));
                const bf16* bp = &Ks[nt * 16 + l15][kk * 32];
                v16bf bb = mk16(*(const v8bf*)(bp + kh * 8), *(const v8bf*)(bp + 16 + kh * 8));
                s[nt] = __builtin_amdgcn_wmma_f32_16x16x32_bf16(
                    false, a, false, bb, (short)0, s[nt], false, false);
            }
        }

        // ---- causal mask + online softmax (16-lane row reductions) ----
        #pragma unroll
        for (int r = 0; r < 8; ++r) {
            const int lrow = wq + r + 8 * kh;            // local q row
            const int qr   = qb * BQ + lrow;             // global q row
            float mx = NEG_BIG;
            float pv[4];
            #pragma unroll
            for (int nt = 0; nt < 4; ++nt) {
                float sv = s[nt][r];
                if (kb0 + nt * 16 + l15 > qr) sv = NEG_BIG;   // causal
                s[nt][r] = sv;
                mx = fmaxf(mx, sv);
            }
            mx = fmaxf(mx, __shfl_xor(mx, 8, 32));
            mx = fmaxf(mx, __shfl_xor(mx, 4, 32));
            mx = fmaxf(mx, __shfl_xor(mx, 2, 32));
            mx = fmaxf(mx, __shfl_xor(mx, 1, 32));
            const float mnew = fmaxf(mstat[r], mx);
            const float corr = __expf(mstat[r] - mnew);
            float rs = 0.0f;
            #pragma unroll
            for (int nt = 0; nt < 4; ++nt) {
                float p = __expf(s[nt][r] - mnew);
                pv[nt] = p; rs += p;
            }
            rs += __shfl_xor(rs, 8, 32);
            rs += __shfl_xor(rs, 4, 32);
            rs += __shfl_xor(rs, 2, 32);
            rs += __shfl_xor(rs, 1, 32);
            lstat[r] = lstat[r] * corr + rs;
            mstat[r] = mnew;
            #pragma unroll
            for (int dt = 0; dt < 4; ++dt) o[dt][r] *= corr;
            #pragma unroll
            for (int nt = 0; nt < 4; ++nt)
                Ps[lrow][nt * 16 + l15] = f2bf(pv[nt]);
        }
        __syncthreads();

        // ---- O += P @ V ----
        #pragma unroll
        for (int dt = 0; dt < 4; ++dt) {
            #pragma unroll
            for (int kk = 0; kk < 2; ++kk) {
                const bf16* ap = &Ps[wq + l15][kk * 32];
                v16bf a = mk16(*(const v8bf*)(ap + kh * 8),
                               *(const v8bf*)(ap + 16 + kh * 8));
                const bf16* bp = &Vs[dt * 16 + l15][kk * 32];
                v16bf bb = mk16(*(const v8bf*)(bp + kh * 8), *(const v8bf*)(bp + 16 + kh * 8));
                o[dt] = __builtin_amdgcn_wmma_f32_16x16x32_bf16(
                    false, a, false, bb, (short)0, o[dt], false, false);
            }
        }
        __syncthreads();
    }

    // ---- epilogue: atty[b, t, h, d] as bf16 (feeds out-proj GEMM A) ----
    #pragma unroll
    for (int r = 0; r < 8; ++r) {
        const float inv = 1.0f / lstat[r];
        const int tq = qb * BQ + wq + r + 8 * kh;
        bf16* dst = atty + (size_t)(b * NT + tq) * NC + h * NHS;
        #pragma unroll
        for (int dt = 0; dt < 4; ++dt)
            dst[dt * 16 + l15] = f2bf(o[dt][r] * inv);
    }
}

// ---------------------------------------------------------------------------
// Host driver
// ---------------------------------------------------------------------------
extern "C" void kernel_launch(void* const* d_in, const int* in_sizes, int n_in,
                              void* d_out, int out_size, void* d_ws, size_t ws_size,
                              hipStream_t stream) {
    (void)in_sizes; (void)n_in; (void)out_size; (void)ws_size;

    const int*   inp  = (const int*)  d_in[0];
    const float* wte  = (const float*)d_in[1];
    const float* wpe  = (const float*)d_in[2];
    const float* ln1g = (const float*)d_in[3];
    const float* ln1b = (const float*)d_in[4];
    const float* wqkv = (const float*)d_in[5];
    const float* bqkv = (const float*)d_in[6];
    const float* wout = (const float*)d_in[7];
    const float* bout = (const float*)d_in[8];
    const float* ln2g = (const float*)d_in[9];
    const float* ln2b = (const float*)d_in[10];
    const float* wfc1 = (const float*)d_in[11];
    const float* bfc1 = (const float*)d_in[12];
    const float* wfc2 = (const float*)d_in[13];
    const float* bfc2 = (const float*)d_in[14];
    const float* lnfg = (const float*)d_in[15];
    const float* lnfb = (const float*)d_in[16];

    char* ws = (char*)d_ws;
    size_t off = 0;
    auto take = [&](size_t bytes) -> char* {
        char* p = ws + off;
        off += (bytes + 255) & ~(size_t)255;
        return p;
    };
    bf16*  wte_bf  = (bf16*)take((size_t)NV * NC * 2);
    bf16*  wqkv_bf = (bf16*)take((size_t)NL * NC * NC3 * 2);
    bf16*  wout_bf = (bf16*)take((size_t)NL * NC * NC * 2);
    bf16*  wfc1_bf = (bf16*)take((size_t)NL * NC * NC4 * 2);
    bf16*  wfc2_bf = (bf16*)take((size_t)NL * NC4 * NC * 2);
    float* x       = (float*)take((size_t)NBT * NC * 4);
    bf16*  hbf     = (bf16*)take((size_t)NBT * NC * 2);
    float* qkvbuf  = (float*)take((size_t)NBT * NC3 * 4);
    bf16*  attybf  = (bf16*)take((size_t)NBT * NC * 2);
    bf16*  actbf   = (bf16*)take((size_t)NBT * NC4 * 2);

    // weights -> bf16 (once per launch; ~250MB, trivial at 23.3 TB/s)
    k_cvt_bf16<<<4096, 256, 0, stream>>>(wte,  wte_bf,  (size_t)NV * NC);
    k_cvt_bf16<<<4096, 256, 0, stream>>>(wqkv, wqkv_bf, (size_t)NL * NC * NC3);
    k_cvt_bf16<<<4096, 256, 0, stream>>>(wout, wout_bf, (size_t)NL * NC * NC);
    k_cvt_bf16<<<4096, 256, 0, stream>>>(wfc1, wfc1_bf, (size_t)NL * NC * NC4);
    k_cvt_bf16<<<4096, 256, 0, stream>>>(wfc2, wfc2_bf, (size_t)NL * NC4 * NC);

    // embeddings
    k_embed<<<NBT, 256, 0, stream>>>(inp, wte, wpe, x);

    const dim3 blk(256);
    for (int l = 0; l < NL; ++l) {
        // ln1 -> h
        k_layernorm<<<NBT, 256, 0, stream>>>(x, ln1g + l * NC, ln1b + l * NC, hbf);
        // qkv = h @ Wqkv + b
        k_gemm<false, false, false, false>
            <<<dim3(NC3 / 128, NBT / 128), blk, 0, stream>>>(
                hbf, wqkv_bf + (size_t)l * NC * NC3, bqkv + (size_t)l * NC3,
                nullptr, qkvbuf, NBT, NC3, NC);
        // fused causal attention
        k_attention<<<dim3(NT / 64, NH, NB), dim3(128), 0, stream>>>(qkvbuf, attybf);
        // x = x + atty @ Wout + b
        k_gemm<false, false, true, false>
            <<<dim3(NC / 128, NBT / 128), blk, 0, stream>>>(
                attybf, wout_bf + (size_t)l * NC * NC, bout + (size_t)l * NC,
                x, x, NBT, NC, NC);
        // ln2 -> h
        k_layernorm<<<NBT, 256, 0, stream>>>(x, ln2g + l * NC, ln2b + l * NC, hbf);
        // act = gelu(h @ Wfc1 + b)  (bf16 out)
        k_gemm<false, true, false, true>
            <<<dim3(NC4 / 128, NBT / 128), blk, 0, stream>>>(
                hbf, wfc1_bf + (size_t)l * NC * NC4, bfc1 + (size_t)l * NC4,
                nullptr, actbf, NBT, NC4, NC);
        // x = x + act @ Wfc2 + b
        k_gemm<false, false, true, false>
            <<<dim3(NC / 128, NBT / 128), blk, 0, stream>>>(
                actbf, wfc2_bf + (size_t)l * NC4 * NC, bfc2 + (size_t)l * NC,
                x, x, NBT, NC, NC4);
    }

    // final LN + weight-tied head: logits = lnf(x) @ wte^T
    k_layernorm<<<NBT, 256, 0, stream>>>(x, lnfg, lnfb, hbf);
    k_gemm<true, false, false, false>
        <<<dim3((NV + 127) / 128, NBT / 128), blk, 0, stream>>>(
            hbf, wte_bf, nullptr, nullptr, (float*)d_out, NBT, NV, NC);
}
